// LearningToDownsample_53584011985479
// MI455X (gfx1250) — 8-bit matrix operand order repaired and checked by reading, compiled, not run on hardware
//
#include <hip/hip_runtime.h>
#include <stdint.h>
#include <stddef.h>

typedef __attribute__((ext_vector_type(8))) int v8i;

namespace off {
constexpr size_t QW1   = 0;
constexpr size_t QDW1  = 3584;
constexpr size_t QDW2  = 4864;
constexpr size_t A1    = 6656;
constexpr size_t A2    = 9728;
constexpr size_t SW3   = 13824;
constexpr size_t SW5   = 14080;
constexpr size_t INV1  = 14336;
constexpr size_t BETA1 = 14464;
constexpr size_t INV2  = 14592;
constexpr size_t BETA2 = 14720;
constexpr size_t INV3  = 14848;
constexpr size_t BETA3 = 15040;
constexpr size_t INV4  = 15232;
constexpr size_t BETA4 = 15424;
constexpr size_t INV5  = 15616;
constexpr size_t BETA5 = 15872;
constexpr size_t ACT1  = 16384;
constexpr size_t ACT2  = 33374464;
constexpr size_t ACT3  = 41763072;
constexpr size_t ACT4  = 54345984;
}

__device__ inline void quant_f32_row(const float* w, int n, float* out) {
  float mx = 0.f;
  for (int i = 0; i < n; ++i) mx = fmaxf(mx, fabsf(w[i]));
  float s = fmaxf(mx / 7.0f, 1e-8f);
  for (int i = 0; i < n; ++i)
    out[i] = fminf(fmaxf(rintf(w[i] / s), -7.f), 7.f) * s;
}

__device__ inline void quant_i8_row(const float* w, int n, int8_t* out64, float* sp) {
  float mx = 0.f;
  for (int i = 0; i < n; ++i) mx = fmaxf(mx, fabsf(w[i]));
  float s = fmaxf(mx / 7.0f, 1e-8f);
  *sp = s;
  for (int i = 0; i < 64; ++i)
    out64[i] = (i < n) ? (int8_t)(int)fminf(fmaxf(rintf(w[i] / s), -7.f), 7.f)
                       : (int8_t)0;
}

__device__ inline void bn_fold(const float* g, const float* b, const float* m,
                               const float* v, int c, float* inv, float* beta) {
  float iv = g[c] / sqrtf(v[c] + 1e-5f);
  inv[c] = iv;
  beta[c] = b[c] - m[c] * iv;
}

__global__ void prep_kernel(
    const float* w1, const float* wdw1, const float* wpw1,
    const float* wdw2, const float* wpw2,
    const float* g1, const float* b1, const float* m1, const float* v1,
    const float* g2, const float* b2, const float* m2, const float* v2,
    const float* g3, const float* b3, const float* m3, const float* v3,
    const float* g4, const float* b4, const float* m4, const float* v4,
    const float* g5, const float* b5, const float* m5, const float* v5,
    char* ws) {
  int t = threadIdx.x;
  float* qw1  = (float*)(ws + off::QW1);
  float* qdw1 = (float*)(ws + off::QDW1);
  float* qdw2 = (float*)(ws + off::QDW2);
  int8_t* A1  = (int8_t*)(ws + off::A1);
  int8_t* A2  = (int8_t*)(ws + off::A2);
  float* sw3  = (float*)(ws + off::SW3);
  float* sw5  = (float*)(ws + off::SW5);

  if (t < 32)        quant_f32_row(w1   + t * 27,        27, qw1  + t * 27);
  else if (t < 64)   quant_f32_row(wdw1 + (t - 32) * 9,   9, qdw1 + (t - 32) * 9);
  else if (t < 112)  quant_i8_row (wpw1 + (t - 64) * 32, 32, A1 + (t - 64) * 64, sw3 + (t - 64));
  else if (t < 160)  quant_f32_row(wdw2 + (t - 112) * 9,  9, qdw2 + (t - 112) * 9);
  else if (t < 224)  quant_i8_row (wpw2 + (t - 160) * 48, 48, A2 + (t - 160) * 64, sw5 + (t - 160));

  if (t < 32)        bn_fold(g1, b1, m1, v1, t,        (float*)(ws + off::INV1), (float*)(ws + off::BETA1));
  else if (t < 64)   bn_fold(g2, b2, m2, v2, t - 32,   (float*)(ws + off::INV2), (float*)(ws + off::BETA2));
  else if (t < 112)  bn_fold(g3, b3, m3, v3, t - 64,   (float*)(ws + off::INV3), (float*)(ws + off::BETA3));
  else if (t < 160)  bn_fold(g4, b4, m4, v4, t - 112,  (float*)(ws + off::INV4), (float*)(ws + off::BETA4));
  else if (t < 224)  bn_fold(g5, b5, m5, v5, t - 160,  (float*)(ws + off::INV5), (float*)(ws + off::BETA5));
}

__global__ void conv1_kernel(const float* __restrict__ x, const float* __restrict__ qw,
                             const float* __restrict__ inv, const float* __restrict__ beta,
                             const float* __restrict__ s1p, uint8_t* __restrict__ out) {
  __shared__ float wsm[32 * 27];
  __shared__ float invs[32], betas[32];
  for (int i = threadIdx.x; i < 32 * 27; i += 256) wsm[i] = qw[i];
  if (threadIdx.x < 32) { invs[threadIdx.x] = inv[threadIdx.x]; betas[threadIdx.x] = beta[threadIdx.x]; }
  __syncthreads();

  int t = blockIdx.x * 256 + threadIdx.x;
  const int TOT = 8 * 255 * 511;
  if (t >= TOT) return;
  int n   = t / 130305;
  int rem = t - n * 130305;
  int h   = rem / 511;
  int w   = rem - h * 511;
  const float s1 = *s1p;

  const float* xb = x + (size_t)n * 3 * 524288 + (size_t)(2 * h) * 1024 + 2 * w;
  float in[27];
#pragma unroll
  for (int ic = 0; ic < 3; ++ic)
#pragma unroll
    for (int r = 0; r < 3; ++r)
#pragma unroll
      for (int c = 0; c < 3; ++c)
        in[ic * 9 + r * 3 + c] = xb[(size_t)ic * 524288 + r * 1024 + c];

  for (int oc = 0; oc < 32; ++oc) {
    float acc = 0.f;
#pragma unroll
    for (int k = 0; k < 27; ++k) acc = fmaf(wsm[oc * 27 + k], in[k], acc);
    float y = acc * invs[oc] + betas[oc];
    float q = fminf(fmaxf(rintf(y / s1), 0.f), 15.f);
    out[(size_t)(n * 32 + oc) * 130305 + rem] = (uint8_t)q;
  }
}

template <int C, int HIN, int WIN, int HOUT, int WOUT>
__global__ void dw_kernel(const uint8_t* __restrict__ in, const float* __restrict__ qw,
                          const float* __restrict__ inv, const float* __restrict__ beta,
                          const float* __restrict__ sinp, const float* __restrict__ soutp,
                          uint8_t* __restrict__ out) {
  int t = blockIdx.x * blockDim.x + threadIdx.x;
  int w0 = t % WOUT;
  int h0 = (t / WOUT) % HOUT;
  int c  = (t / (WOUT * HOUT)) % C;
  int n  = t / (WOUT * HOUT * C);
  const float s_in = *sinp, s_out = *soutp;

  const uint8_t* plane = in + (size_t)(n * C + c) * (HIN * WIN);
  float wl[9];
#pragma unroll
  for (int i = 0; i < 9; ++i) wl[i] = qw[c * 9 + i];

  float acc = 0.f;
#pragma unroll
  for (int kh = 0; kh < 3; ++kh) {
    int hi = 2 * h0 - 1 + kh;
#pragma unroll
    for (int kw = 0; kw < 3; ++kw) {
      int wi = 2 * w0 - 1 + kw;
      if (hi >= 0 && hi < HIN && wi >= 0 && wi < WIN)
        acc += wl[kh * 3 + kw] * (float)plane[hi * WIN + wi];
    }
  }
  float y = acc * s_in * inv[c] + beta[c];
  float q = fminf(fmaxf(rintf(y / s_out), 0.f), 15.f);
  out[t] = (uint8_t)q;
}

template <int CIN, int COUT, int HW, bool FINAL>
__global__ void pw_kernel(const uint8_t* __restrict__ X, const int8_t* __restrict__ A8,
                          const float* __restrict__ sw, const float* __restrict__ inv,
                          const float* __restrict__ beta,
                          const float* __restrict__ sinp, const float* __restrict__ soutp,
                          uint8_t* __restrict__ outLv, float* __restrict__ outF) {
  const int lane  = threadIdx.x & 31;
  const int wave  = blockIdx.x * (blockDim.x >> 5) + (threadIdx.x >> 5);
  const int pbase = wave << 4;
  const int n     = pbase / HW;
  const int p     = pbase % HW;
  const int Ncol  = lane & 15;
  const int kh    = lane >> 4;
  const float s_in = *sinp, s_out = *soutp;

  const uint8_t* xp = X + (size_t)n * CIN * HW + (p + Ncol);
  v8i B = {0, 0, 0, 0, 0, 0, 0, 0};
#pragma unroll
  for (int r = 0; r < 8; ++r) {
    int kb = kh * 8 + (r >> 1) * 16 + (r & 1) * 4;
    unsigned d = 0;
#pragma unroll
    for (int b = 0; b < 4; ++b) {
      int k = kb + b;
      unsigned byte = (k < CIN) ? (unsigned)xp[(size_t)k * HW] : 0u;
      d |= byte << (8 * b);
    }
    B[r] = (int)d;
  }

#pragma unroll
  for (int ct = 0; ct < COUT / 16; ++ct) {
    int row = ct * 16 + (lane & 15);
    const int8_t* ap = A8 + row * 64 + kh * 8;
    v8i A;
#pragma unroll
    for (int j = 0; j < 4; ++j) {
      A[2 * j]     = *(const int*)(ap + 16 * j);
      A[2 * j + 1] = *(const int*)(ap + 16 * j + 4);
    }
    v8i Cacc = {0, 0, 0, 0, 0, 0, 0, 0};
    Cacc = __builtin_amdgcn_wmma_i32_16x16x64_iu8(
         true, A,  false, B, Cacc,
         false,  false);

#pragma unroll
    for (int r = 0; r < 8; ++r) {
      int c = ct * 16 + r + kh * 8;
      float y = (float)Cacc[r] * (s_in * sw[c]);
      y = y * inv[c] + beta[c];
      float q = fminf(fmaxf(rintf(y / s_out), 0.f), 15.f);
      size_t oidx = (size_t)(n * COUT + c) * HW + (p + Ncol);
      if (FINAL) outF[oidx] = q * s_out;
      else       outLv[oidx] = (uint8_t)q;
    }
  }
}

extern "C" void kernel_launch(void* const* d_in, const int* in_sizes, int n_in,
                              void* d_out, int out_size, void* d_ws, size_t ws_size,
                              hipStream_t stream) {
  const float* x    = (const float*)d_in[0];
  const float* w1   = (const float*)d_in[1];
  const float* g1   = (const float*)d_in[2];
  const float* b1   = (const float*)d_in[3];
  const float* m1   = (const float*)d_in[4];
  const float* v1   = (const float*)d_in[5];
  const float* s1   = (const float*)d_in[6];
  const float* wdw1 = (const float*)d_in[7];
  const float* g2   = (const float*)d_in[8];
  const float* b2   = (const float*)d_in[9];
  const float* m2   = (const float*)d_in[10];
  const float* v2   = (const float*)d_in[11];
  const float* s2   = (const float*)d_in[12];
  const float* wpw1 = (const float*)d_in[13];
  const float* g3   = (const float*)d_in[14];
  const float* b3   = (const float*)d_in[15];
  const float* m3   = (const float*)d_in[16];
  const float* v3   = (const float*)d_in[17];
  const float* s3   = (const float*)d_in[18];
  const float* wdw2 = (const float*)d_in[19];
  const float* g4   = (const float*)d_in[20];
  const float* b4   = (const float*)d_in[21];
  const float* m4   = (const float*)d_in[22];
  const float* v4   = (const float*)d_in[23];
  const float* s4   = (const float*)d_in[24];
  const float* wpw2 = (const float*)d_in[25];
  const float* g5   = (const float*)d_in[26];
  const float* b5   = (const float*)d_in[27];
  const float* m5   = (const float*)d_in[28];
  const float* v5   = (const float*)d_in[29];
  const float* s5   = (const float*)d_in[30];

  char* ws = (char*)d_ws;
  float*   qw1  = (float*)(ws + off::QW1);
  float*   qdw1 = (float*)(ws + off::QDW1);
  float*   qdw2 = (float*)(ws + off::QDW2);
  int8_t*  A1   = (int8_t*)(ws + off::A1);
  int8_t*  A2   = (int8_t*)(ws + off::A2);
  float*   sw3  = (float*)(ws + off::SW3);
  float*   sw5  = (float*)(ws + off::SW5);
  float*   inv1 = (float*)(ws + off::INV1);
  float*   bt1  = (float*)(ws + off::BETA1);
  float*   inv2 = (float*)(ws + off::INV2);
  float*   bt2  = (float*)(ws + off::BETA2);
  float*   inv3 = (float*)(ws + off::INV3);
  float*   bt3  = (float*)(ws + off::BETA3);
  float*   inv4 = (float*)(ws + off::INV4);
  float*   bt4  = (float*)(ws + off::BETA4);
  float*   inv5 = (float*)(ws + off::INV5);
  float*   bt5  = (float*)(ws + off::BETA5);
  uint8_t* act1 = (uint8_t*)(ws + off::ACT1);
  uint8_t* act2 = (uint8_t*)(ws + off::ACT2);
  uint8_t* act3 = (uint8_t*)(ws + off::ACT3);
  uint8_t* act4 = (uint8_t*)(ws + off::ACT4);

  prep_kernel<<<1, 256, 0, stream>>>(
      w1, wdw1, wpw1, wdw2, wpw2,
      g1, b1, m1, v1, g2, b2, m2, v2, g3, b3, m3, v3,
      g4, b4, m4, v4, g5, b5, m5, v5, ws);

  conv1_kernel<<<4073, 256, 0, stream>>>(x, qw1, inv1, bt1, s1, act1);

  dw_kernel<32, 255, 511, 128, 256><<<32768, 256, 0, stream>>>(
      act1, qdw1, inv2, bt2, s1, s2, act2);

  pw_kernel<32, 48, 32768, false><<<2048, 256, 0, stream>>>(
      act2, A1, sw3, inv3, bt3, s2, s3, act3, nullptr);

  dw_kernel<48, 128, 256, 64, 128><<<12288, 256, 0, stream>>>(
      act3, qdw2, inv4, bt4, s3, s4, act4);

  pw_kernel<48, 64, 8192, true><<<512, 256, 0, stream>>>(
      act4, A2, sw5, inv5, bt5, s4, s5, nullptr, (float*)d_out);
}
